// smGNN_32839319945332
// MI455X (gfx1250) — matrix operand order repaired and checked by reading, not compiled, not run on hardware
//
#include <hip/hip_runtime.h>
#include <hip/hip_bf16.h>

// Problem constants (match reference)
#define Bb   32
#define Nn   10000
#define Ee   160000
#define Hh   32
#define Cc   16
#define NFf  2000
#define EEc  (Ee + Nn)          // edges + self loops = 170000
#define BN   (Bb * Nn)          // 320000
#define BNH  (BN * Hh)          // 10240000
#define EPSc 1e-5f

typedef __attribute__((ext_vector_type(16))) _Float16 v16h;
typedef __attribute__((ext_vector_type(8)))  float    v8f;

// ---- order-preserving float <-> uint map for atomicMax on floats ----
__device__ __forceinline__ unsigned f2ord(float f) {
    unsigned b = __float_as_uint(f);
    return (b & 0x80000000u) ? ~b : (b | 0x80000000u);
}
__device__ __forceinline__ float ord2f(unsigned u) {
    return __uint_as_float((u & 0x80000000u) ? (u & 0x7FFFFFFFu) : ~u);
}
#define ORD_NEG_INF 0x007FFFFFu   // f2ord(-inf)

// ---------------- fills ----------------
__global__ void k_fill_f32(float* p, float v, int n) {
    int i = blockIdx.x * blockDim.x + threadIdx.x;
    if (i < n) p[i] = v;
}
__global__ void k_fill_u32(unsigned* p, unsigned v, int n) {
    int i = blockIdx.x * blockDim.x + threadIdx.x;
    if (i < n) p[i] = v;
}
__global__ void k_fill_f64(double* p, double v, int n) {
    int i = blockIdx.x * blockDim.x + threadIdx.x;
    if (i < n) p[i] = v;
}

// ---------------- layer 1: h = mask(x) @ W1 ; s,d scalars (F_IN == 1) ----------------
__global__ void k_layer1(const float* __restrict__ x, const unsigned char* __restrict__ mask,
                         const float* __restrict__ W1, const float* __restrict__ as1,
                         const float* __restrict__ ad1,
                         float* __restrict__ h, float* __restrict__ s, float* __restrict__ d)
{
    int tid = blockIdx.x * blockDim.x + threadIdx.x;
    int node = tid >> 5;        // b*N + n
    int f    = tid & 31;
    if (node >= BN) return;
    int n = node % Nn;
    float w  = W1[f];
    float xm = mask[n] ? 0.0f : x[node];
    h[(size_t)node * Hh + f] = xm * w;
    float sa = w * as1[f];
    float da = w * ad1[f];
    #pragma unroll
    for (int m = 16; m >= 1; m >>= 1) {
        sa += __shfl_xor(sa, m, 32);
        da += __shfl_xor(da, m, 32);
    }
    if (f == 0) { s[node] = xm * sa; d[node] = xm * da; }
}

// ---------------- s,d from dense h (layer 2) ----------------
__global__ void k_sd(const float* __restrict__ h, const float* __restrict__ av,
                     const float* __restrict__ dv, float* __restrict__ s, float* __restrict__ d)
{
    int tid = blockIdx.x * blockDim.x + threadIdx.x;
    int node = tid >> 5;
    int f    = tid & 31;
    if (node >= BN) return;
    float v = h[(size_t)node * Hh + f];
    float sa = v * av[f];
    float da = v * dv[f];
    #pragma unroll
    for (int m = 16; m >= 1; m >>= 1) {
        sa += __shfl_xor(sa, m, 32);
        da += __shfl_xor(da, m, 32);
    }
    if (f == 0) { s[node] = sa; d[node] = da; }
}

__device__ __forceinline__ void edge_sd(const int* __restrict__ ei, int e, int& src, int& dst) {
    if (e < Ee) { src = ei[e]; dst = ei[Ee + e]; }
    else        { src = e - Ee; dst = e - Ee; }
}
__device__ __forceinline__ float leaky(float v) { return v > 0.0f ? v : 0.2f * v; }

// ---------------- edge pass 1: segment max ----------------
__global__ void k_edge_max(const int* __restrict__ ei, const float* __restrict__ s,
                           const float* __restrict__ d, unsigned* __restrict__ m)
{
    int tid = blockIdx.x * blockDim.x + threadIdx.x;
    if (tid >= Bb * EEc) return;
    int b = tid / EEc, e = tid % EEc;
    int src, dst; edge_sd(ei, e, src, dst);
    float ev = leaky(s[b * Nn + src] + d[b * Nn + dst]);
    atomicMax(&m[b * Nn + dst], f2ord(ev));
}

// ---------------- edge pass 2: segment sum of exp ----------------
__global__ void k_edge_sum(const int* __restrict__ ei, const float* __restrict__ s,
                           const float* __restrict__ d, const unsigned* __restrict__ m,
                           float* __restrict__ den)
{
    int tid = blockIdx.x * blockDim.x + threadIdx.x;
    if (tid >= Bb * EEc) return;
    int b = tid / EEc, e = tid % EEc;
    int src, dst; edge_sd(ei, e, src, dst);
    float ev = leaky(s[b * Nn + src] + d[b * Nn + dst]);
    float ex = __expf(ev - ord2f(m[b * Nn + dst]));
    atomicAdd(&den[b * Nn + dst], ex);
}

// ---------------- edge pass 3: weighted aggregation (1 lane per feature) ----------------
__global__ void k_edge_agg(const int* __restrict__ ei, const float* __restrict__ s,
                           const float* __restrict__ d, const unsigned* __restrict__ m,
                           const float* __restrict__ den, const float* __restrict__ h,
                           float* __restrict__ out)
{
    int tid  = blockIdx.x * blockDim.x + threadIdx.x;
    int lane = tid & 31;
    int idx  = tid >> 5;
    if (idx >= Bb * EEc) return;
    int b = idx / EEc, e = idx % EEc;
    int src, dst; edge_sd(ei, e, src, dst);
    int bd = b * Nn + dst;
    float ev = leaky(s[b * Nn + src] + d[bd]);
    float alpha = __expf(ev - ord2f(m[bd])) / den[bd];
    atomicAdd(&out[(size_t)bd * Hh + lane],
              alpha * h[((size_t)b * Nn + src) * Hh + lane]);
}

// ---------------- add bias + accumulate global-LN stats ----------------
__global__ void k_bias_stats(float* __restrict__ h, const float* __restrict__ bias,
                             double* __restrict__ stats)
{
    __shared__ double ss[256], ss2[256];
    double ls = 0.0, ls2 = 0.0;
    for (int i = blockIdx.x * blockDim.x + threadIdx.x; i < BNH; i += gridDim.x * blockDim.x) {
        float v = h[i] + bias[i & 31];
        h[i] = v;
        ls  += (double)v;
        ls2 += (double)v * (double)v;
    }
    ss[threadIdx.x] = ls; ss2[threadIdx.x] = ls2;
    __syncthreads();
    for (int st = 128; st > 0; st >>= 1) {
        if (threadIdx.x < st) { ss[threadIdx.x] += ss[threadIdx.x + st]; ss2[threadIdx.x] += ss2[threadIdx.x + st]; }
        __syncthreads();
    }
    if (threadIdx.x == 0) { atomicAdd(&stats[0], ss[0]); atomicAdd(&stats[1], ss2[0]); }
}

// ---------------- graph-LN + ELU, emit f16 for WMMA ----------------
__global__ void k_ln_elu(const float* __restrict__ h, const double* __restrict__ stats,
                         const float* __restrict__ g, const float* __restrict__ be,
                         _Float16* __restrict__ out)
{
    int i = blockIdx.x * blockDim.x + threadIdx.x;
    if (i >= BNH) return;
    float mu  = (float)(stats[0] / (double)BNH);
    float var = (float)(stats[1] / (double)BNH) - mu * mu;
    float sd  = sqrtf(fmaxf(var, 0.0f));
    float inv = 1.0f / (sd + EPSc);
    int f = i & 31;
    float y = (h[i] - mu) * inv * g[f] + be[f];
    y = y > 0.0f ? y : (__expf(y) - 1.0f);
    out[i] = (_Float16)y;
}

// ---------------- WMMA GEMM: OUT[BN x NCOLS] = A(f16, BN x 32) @ W(f32, 32 x NCOLS) ----------------
// One wave computes a 16-row tile; K=32 exactly matches v_wmma_f32_16x16x32_f16.
template <int NCOLS, bool FINAL>
__global__ void k_gemm(const _Float16* __restrict__ Ah, const float* __restrict__ W,
                       const float* __restrict__ bias, float* __restrict__ out)
{
    int wave = (blockIdx.x * blockDim.x + threadIdx.x) >> 5;
    int lane = threadIdx.x & 31;
    int r0   = wave * 16;
    if (r0 >= BN) return;

    // A layout (16-bit A 16x32): lanes 0-15 -> M=lane, K{0..7,16..23}; lanes 16-31 -> M=lane-16, K{8..15,24..31}
    const _Float16* arow = Ah + (size_t)(r0 + (lane & 15)) * Hh;
    int kb = (lane < 16) ? 0 : 8;
    v16h a;
    #pragma unroll
    for (int j = 0; j < 8; j++) { a[j] = arow[kb + j]; a[8 + j] = arow[16 + kb + j]; }

    // B layout (16-bit B 32x16): element j of v16h = row (j<8 ? kb+j : 16+kb+j-8), column lane&15
    int col = lane & 15;
    int kr  = (lane < 16) ? 0 : 16;
    int mA  = (lane < 16) ? 0 : 8;

    #pragma unroll
    for (int nt = 0; nt < NCOLS / 16; nt++) {
        v16h bm;
        #pragma unroll
        for (int j = 0; j < 16; j++) bm[j] = (_Float16)W[((j < 8) ? (kb + j) : (16 + kb + j - 8)) * NCOLS + nt * 16 + col];
        v8f c = {};
        c = __builtin_amdgcn_wmma_f32_16x16x32_f16(false, a, false, bm, (short)0, c, false, false);
        int nO = nt * 16 + col;
        #pragma unroll
        for (int i = 0; i < 8; i++) {
            float v = c[i];
            if (FINAL) { v += bias[nO]; v = 1.0f / (1.0f + __expf(-v)); }
            out[(size_t)(r0 + mA + i) * NCOLS + nO] = v;
        }
    }
}

// ---------------- final gather + transpose ----------------
__global__ void k_gather(const float* __restrict__ probs, const int* __restrict__ fn,
                         float* __restrict__ out)
{
    int tid = blockIdx.x * blockDim.x + threadIdx.x;
    if (tid >= Bb * Cc * NFf) return;
    int b   = tid / (Cc * NFf);
    int rem = tid % (Cc * NFf);
    int c   = rem / NFf;
    int i   = rem % NFf;
    int node = fn[i];
    out[tid] = probs[((size_t)b * Nn + node) * Cc + c];
}

extern "C" void kernel_launch(void* const* d_in, const int* in_sizes, int n_in,
                              void* d_out, int out_size, void* d_ws, size_t ws_size,
                              hipStream_t stream)
{
    const float*         x    = (const float*)d_in[0];
    const int*           ei   = (const int*)d_in[1];
    const int*           fn   = (const int*)d_in[2];
    const unsigned char* mask = (const unsigned char*)d_in[3];
    const float* W1   = (const float*)d_in[4];
    const float* as1  = (const float*)d_in[5];
    const float* ad1  = (const float*)d_in[6];
    const float* b1   = (const float*)d_in[7];
    const float* g1   = (const float*)d_in[8];
    const float* be1  = (const float*)d_in[9];
    const float* W2   = (const float*)d_in[10];
    const float* as2  = (const float*)d_in[11];
    const float* ad2  = (const float*)d_in[12];
    const float* b2   = (const float*)d_in[13];
    const float* g2   = (const float*)d_in[14];
    const float* be2  = (const float*)d_in[15];
    const float* Wlin = (const float*)d_in[16];
    const float* blin = (const float*)d_in[17];

    // workspace carve-out (all sizes multiples of 8 bytes)
    float*     h_src  = (float*)d_ws;                    // BNH f32  (h1 / h2)
    float*     h_agg  = h_src + BNH;                     // BNH f32  (aggregation / biased)
    _Float16*  h_half = (_Float16*)(h_agg + BNH);        // BNH f16  (LN+ELU output for WMMA)
    float*     sbuf   = (float*)(h_half + BNH);          // BN
    float*     dbuf   = sbuf + BN;                       // BN
    unsigned*  mbuf   = (unsigned*)(dbuf + BN);          // BN
    float*     den    = (float*)(mbuf + BN);             // BN
    double*    stats  = (double*)(den + BN);             // 2 doubles
    float*     probs  = h_src;                           // reuse dead h2 buffer (BN*Cc)

    const int TB = 256;
    const int gBN32  = (BN * 32 + TB - 1) / TB;          // node-wave kernels
    const int gBNH   = (BNH + TB - 1) / TB;
    const int gBE    = (Bb * EEc + TB - 1) / TB;         // per-edge
    const int gBE32  = (Bb * EEc * 32 + TB - 1) / TB;    // per-edge-feature
    const int gGEMM  = ((BN / 16) * 32 + TB - 1) / TB;   // BN/16 waves
    const int gBNfill = (BN + TB - 1) / TB;
    const int gOUT   = (Bb * Cc * NFf + TB - 1) / TB;

    for (int layer = 0; layer < 2; layer++) {
        // reset segment buffers + stats
        k_fill_u32<<<gBNfill, TB, 0, stream>>>(mbuf, ORD_NEG_INF, BN);
        k_fill_f32<<<gBNfill, TB, 0, stream>>>(den, 0.0f, BN);
        k_fill_f32<<<gBNH,    TB, 0, stream>>>(h_agg, 0.0f, BNH);
        k_fill_f64<<<1, 64,      0, stream>>>(stats, 0.0, 2);

        if (layer == 0) {
            k_layer1<<<gBN32, TB, 0, stream>>>(x, mask, W1, as1, ad1, h_src, sbuf, dbuf);
        } else {
            k_gemm<32, false><<<gGEMM, TB, 0, stream>>>(h_half, W2, W2, h_src);
            k_sd<<<gBN32, TB, 0, stream>>>(h_src, as2, ad2, sbuf, dbuf);
        }

        k_edge_max<<<gBE,   TB, 0, stream>>>(ei, sbuf, dbuf, mbuf);
        k_edge_sum<<<gBE,   TB, 0, stream>>>(ei, sbuf, dbuf, mbuf, den);
        k_edge_agg<<<gBE32, TB, 0, stream>>>(ei, sbuf, dbuf, mbuf, den, h_src, h_agg);

        const float* bias = (layer == 0) ? b1 : b2;
        const float* g    = (layer == 0) ? g1 : g2;
        const float* be   = (layer == 0) ? be1 : be2;
        k_bias_stats<<<4096, TB, 0, stream>>>(h_agg, bias, stats);
        k_ln_elu<<<gBNH, TB, 0, stream>>>(h_agg, stats, g, be, h_half);
    }

    // final linear (32 -> 16) + sigmoid via WMMA, then gather/transpose
    k_gemm<16, true><<<gGEMM, TB, 0, stream>>>(h_half, Wlin, blin, probs);
    k_gather<<<gOUT, TB, 0, stream>>>(probs, fn, (float*)d_out);
}
